// RawMLPConcatPredictor_63599875719679
// MI455X (gfx1250) — matrix operand order repaired and checked by reading, not compiled, not run on hardware
//
#include <hip/hip_runtime.h>

typedef __attribute__((ext_vector_type(16))) __bf16 v16bf;
typedef __attribute__((ext_vector_type(2)))  __bf16 v2bf;
typedef __attribute__((ext_vector_type(8)))  float  v8f;
typedef __attribute__((ext_vector_type(2)))  float  v2f;

#define NEDGES   262144
#define IN_DIM   768
#define KTOT     1536      // 2 * IN_DIM
#define HIDDEN   256
#define M_BLK    64
#define KCHUNK   128
#define A_STRIDE 136       // 128 + 8 bf16 padding (272B row stride, bank-rotated)

// native f32 -> bf16 (RNE) pair conversion; lowers to v_cvt_pk_bf16_f32-class ops
static __device__ __forceinline__ unsigned pack_bf2(float x, float y) {
    v2bf r = __builtin_convertvector((v2f){x, y}, v2bf);
    union { v2bf b; unsigned u; } u; u.b = r;
    return u.u;
}

static __device__ __forceinline__ unsigned short f2bf1(float f) {
    __bf16 b = (__bf16)f;
    union { __bf16 b; unsigned short u; } u; u.b = b;
    return u.u;
}

// ---- prep: W1 [1536][256] f32  ->  W1T [256][1536] bf16 (in d_ws) ----
__global__ void w1_transpose_bf16(const float* __restrict__ W1,
                                  unsigned short* __restrict__ W1T) {
    const int n = threadIdx.x;   // 0..255
    const int k = blockIdx.x;    // 0..1535
    W1T[(size_t)n * KTOT + k] = f2bf1(W1[(size_t)k * HIDDEN + n]);
}

// ---- main: gather + bf16 GEMM + bias/ReLU + W2 reduction ----
__global__ __launch_bounds__(256) void edge_mlp_kernel(
    const float* __restrict__ x_src, const float* __restrict__ x_dst,
    const int* __restrict__ s_idx,  const int* __restrict__ d_idx,
    const unsigned short* __restrict__ W1T,
    const float* __restrict__ b1, const float* __restrict__ W2,
    const float* __restrict__ b2, float* __restrict__ out)
{
    __shared__ unsigned short a_sh[M_BLK * A_STRIDE];
    __shared__ float lds_logits[M_BLK];
    __shared__ int sidx_sh[M_BLK];
    __shared__ int didx_sh[M_BLK];

    const int tid  = threadIdx.x;
    const int e0   = blockIdx.x * M_BLK;
    const int lane = tid & 31;
    const int wave = tid >> 5;
    const int li   = lane & 15;
    const int half = lane >> 4;

    if (tid < M_BLK) {
        sidx_sh[tid] = s_idx[e0 + tid];
        didx_sh[tid] = d_idx[e0 + tid];
        lds_logits[tid] = 0.0f;
    }

    v8f acc[4][2];
#pragma unroll
    for (int mi = 0; mi < 4; ++mi)
#pragma unroll
        for (int nj = 0; nj < 2; ++nj)
            acc[mi][nj] = (v8f){0.f,0.f,0.f,0.f,0.f,0.f,0.f,0.f};

    const int n0 = wave * 32;                 // this wave's 32 hidden columns
    const unsigned short* bptr0 = W1T + (size_t)(n0 + li)      * KTOT + half * 16;
    const unsigned short* bptr1 = W1T + (size_t)(n0 + 16 + li) * KTOT + half * 16;

    const int srow = tid >> 2;  // 0..63 : edge-row staged by this thread
    const int ssub = tid & 3;   // 4 threads/row, 32 floats each

    for (int kc = 0; kc < KTOT / KCHUNK; ++kc) {    // 12 chunks of 128
        __syncthreads();
        // ---- stage gathered f32 rows as bf16 into LDS ----
        {
            const float* src;
            if (kc < IN_DIM / KCHUNK)
                src = x_src + (size_t)sidx_sh[srow] * IN_DIM + kc * KCHUNK + ssub * 32;
            else
                src = x_dst + (size_t)didx_sh[srow] * IN_DIM + (kc - IN_DIM / KCHUNK) * KCHUNK + ssub * 32;
            unsigned short* dst = a_sh + srow * A_STRIDE + ssub * 32;
#pragma unroll
            for (int i = 0; i < 4; ++i) {
                float4 fa = ((const float4*)src)[2 * i];
                float4 fb = ((const float4*)src)[2 * i + 1];
                uint4 p;
                p.x = pack_bf2(fa.x, fa.y);
                p.y = pack_bf2(fa.z, fa.w);
                p.z = pack_bf2(fb.x, fb.y);
                p.w = pack_bf2(fb.z, fb.w);
                ((uint4*)dst)[i] = p;
            }
        }
        __syncthreads();

        // ---- 4 WMMA k-steps of 32 over this chunk ----
#pragma unroll
        for (int ks = 0; ks < 4; ++ks) {
            const int kloc  = ks * 32;
            const int kglob = kc * KCHUNK + kloc;

            v16bf bfr0 = *(const v16bf*)(bptr0 + kglob);   // 32B contiguous per lane
            v16bf bfr1 = *(const v16bf*)(bptr1 + kglob);

            v16bf afr[4];
#pragma unroll
            for (int mi = 0; mi < 4; ++mi) {
                const unsigned short* ap =
                    a_sh + (mi * 16 + li) * A_STRIDE + kloc + half * 16;
                union { uint4 q[2]; v16bf v; } u;
                u.q[0] = *(const uint4*)(ap);        // K = k0..k0+7   (+16 for half)
                u.q[1] = *(const uint4*)(ap + 8);    // K = k0+8..15   (+16 for half)
                afr[mi] = u.v;
            }
#pragma unroll
            for (int mi = 0; mi < 4; ++mi) {
                acc[mi][0] = __builtin_amdgcn_wmma_f32_16x16x32_bf16(
                    false, afr[mi], false, bfr0, (short)0, acc[mi][0], false, false);
                acc[mi][1] = __builtin_amdgcn_wmma_f32_16x16x32_bf16(
                    false, afr[mi], false, bfr1, (short)0, acc[mi][1], false, false);
            }
        }
    }

    // ---- epilogue: + b1, ReLU, * W2, per-edge reduce via LDS atomics ----
    const float b1v0 = b1[n0 + li],      b1v1 = b1[n0 + 16 + li];
    const float w2v0 = W2[n0 + li],      w2v1 = W2[n0 + 16 + li];

#pragma unroll
    for (int mi = 0; mi < 4; ++mi) {
#pragma unroll
        for (int j = 0; j < 8; ++j) {
            float h0 = acc[mi][0][j] + b1v0; h0 = h0 > 0.f ? h0 : 0.f;
            float h1 = acc[mi][1][j] + b1v1; h1 = h1 > 0.f ? h1 : 0.f;
            atomicAdd(&lds_logits[mi * 16 + j + half * 8], h0 * w2v0 + h1 * w2v1);
        }
    }
    __syncthreads();
    if (tid < M_BLK) out[e0 + tid] = lds_logits[tid] + b2[0];
}

extern "C" void kernel_launch(void* const* d_in, const int* in_sizes, int n_in,
                              void* d_out, int out_size, void* d_ws, size_t ws_size,
                              hipStream_t stream) {
    const float* x_src = (const float*)d_in[0];
    const float* x_dst = (const float*)d_in[1];
    const int*   s_idx = (const int*)d_in[2];
    const int*   d_idx = (const int*)d_in[3];
    const float* W1    = (const float*)d_in[4];
    const float* b1    = (const float*)d_in[5];
    const float* W2    = (const float*)d_in[6];
    const float* b2    = (const float*)d_in[7];
    float* out = (float*)d_out;

    unsigned short* W1T = (unsigned short*)d_ws;   // 256*1536*2 = 768 KB

    w1_transpose_bf16<<<KTOT, HIDDEN, 0, stream>>>(W1, W1T);
    edge_mlp_kernel<<<NEDGES / M_BLK, 256, 0, stream>>>(
        x_src, x_dst, s_idx, d_idx, W1T, b1, W2, b2, out);
}
